// ConvNext_85667417686339
// MI455X (gfx1250) — matrix operand order repaired and checked by reading, not compiled, not run on hardware
//
#include <hip/hip_runtime.h>
#include <hip/hip_bf16.h>
#include <math.h>

typedef __attribute__((ext_vector_type(16))) __bf16        v16bf;
typedef __attribute__((ext_vector_type(8)))  __bf16        v8bf;
typedef __attribute__((ext_vector_type(8)))  float         v8f;
typedef __attribute__((ext_vector_type(4)))  float         v4f;
typedef __attribute__((ext_vector_type(4)))  unsigned int  v4u;
typedef __attribute__((ext_vector_type(8)))  int           v8i;
typedef __attribute__((ext_vector_type(4)))  int           v4i;

#define CCH 64   // channels
#define KD  32   // kernel basis dim
#define HH  256  // hidden dim

// Build a v16bf WMMA fragment from two 8-float contiguous chunks (f32 -> bf16).
static __device__ inline v16bf frag_f32(const float* p0, const float* p1) {
  v4f a0 = *(const v4f*)p0, a1 = *(const v4f*)(p0 + 4);
  v4f b0 = *(const v4f*)p1, b1 = *(const v4f*)(p1 + 4);
  v16bf f;
#pragma unroll
  for (int i = 0; i < 4; i++) {
    f[i]      = (__bf16)a0[i];
    f[i + 4]  = (__bf16)a1[i];
    f[i + 8]  = (__bf16)b0[i];
    f[i + 12] = (__bf16)b1[i];
  }
  return f;
}

// Build a v16bf fragment from two 8-bf16 contiguous chunks (LDS or global).
static __device__ inline v16bf frag_bf16(const __bf16* p0, const __bf16* p1) {
  v8bf lo = *(const v8bf*)p0;
  v8bf hi = *(const v8bf*)p1;
  v16bf f;
#pragma unroll
  for (int i = 0; i < 8; i++) { f[i] = lo[i]; f[i + 8] = hi[i]; }
  return f;
}

// ---------------------------------------------------------------------------
// Kernel 0: convert W1 [H,C] and W2 [C,H] f32 -> bf16 into workspace.
// ---------------------------------------------------------------------------
__global__ void cvt_kernel(const float* __restrict__ W1, const float* __restrict__ W2,
                           __bf16* __restrict__ W1b, __bf16* __restrict__ W2b,
                           int n1, int n2) {
  int i = blockIdx.x * blockDim.x + threadIdx.x;
  if (i < n1) W1b[i] = (__bf16)W1[i];
  if (i < n2) W2b[i] = (__bf16)W2[i];
}

// ---------------------------------------------------------------------------
// Kernel 1: per-edge depthwise kernel via WMMA (K=32 exactly), then
// message = kernel * x[src], scatter-add into x1[dst] (f32 atomics, L2-bound).
// One wave handles a 16-edge tile; 4 WMMAs cover the 64 channels.
// Fast path: full tiles take a guard-free, 32-bit-offset scatter sequence;
// gathers are issued in a batch before the atomics so the 8 loads pipeline.
// ---------------------------------------------------------------------------
__global__ __launch_bounds__(256) void edge_kernel(
    const float* __restrict__ x, const float* __restrict__ basis,
    const long long* __restrict__ ei, const float* __restrict__ Wk,
    float* __restrict__ x1, int E) {
  int lane = threadIdx.x & 31, wave = threadIdx.x >> 5;
  int lo = lane & 15, hi = lane >> 4;
  long long e0 = ((long long)blockIdx.x * 8 + wave) * 16;
  if (e0 >= E) return;

  // speculative prefetch of the next block's basis tile -> global_prefetch_b8
  __builtin_prefetch(basis + (e0 + 128) * KD, 0, 1);

  // A fragment: 16 edges x 32 basis coeffs (contiguous k = 16*hi + s)
  long long eA = e0 + lo; if (eA >= (long long)E) eA = E - 1;
  const float* br = basis + eA * KD;
  int k0 = hi * 16;
  v16bf a = frag_f32(br + k0, br + k0 + 8);

  // One edge-index load per lane (row = lane&15); node ids fit in int32.
  // Broadcast the 8 rows each half-lane owns via wave shuffles.
  int sl = (int)ei[eA];
  int dl = (int)ei[(long long)E + eA];
  int soff[8], doff[8];
#pragma unroll
  for (int j = 0; j < 8; j++) {
    int m = j + 8 * hi;
    soff[j] = __shfl(sl, m) * CCH;
    doff[j] = __shfl(dl, m) * CCH;
  }

  bool full = (e0 + 16 <= (long long)E);  // wave-uniform branch

#pragma unroll
  for (int ct = 0; ct < 4; ct++) {
    // B fragment: kernel_W row (= output channel), 16 contiguous f32 K-values
    const float* wr = Wk + (ct * 16 + lo) * KD + hi * 16;
    v16bf b = frag_f32(wr, wr + 8);
    v8f c = {};
    c = __builtin_amdgcn_wmma_f32_16x16x32_bf16(false, a, false, b,
                                                (short)0, c, false, false);
    int ch = ct * 16 + lo;
    if (full) {
      // batch the 8 gathers (64B-coalesced rows), then drain into atomics
      float xv[8];
#pragma unroll
      for (int j = 0; j < 8; j++) xv[j] = x[soff[j] + ch];
#pragma unroll
      for (int j = 0; j < 8; j++)
        atomicAdd(&x1[doff[j] + ch], c[j] * xv[j]);  // 64B-coalesced atomics
    } else {
#pragma unroll
      for (int j = 0; j < 8; j++) {
        if (e0 + j + 8 * hi < (long long)E) {
          float val = c[j] * x[soff[j] + ch];
          atomicAdd(&x1[doff[j] + ch], val);
        }
      }
    }
  }
}

// ---------------------------------------------------------------------------
// Kernel 2: fused  bias + LayerNorm + (64->256 GEMM, exact GELU) +
// (256->64 GEMM) + layer_scale + residual.  128 rows per block (8 waves x 16).
// W1 (bf16, 32KB) is pulled into LDS by the Tensor Data Mover; W2 B-fragments
// stream from L2.  GELU outputs are re-laid-out through per-wave LDS staging.
// x1 lives in d_out and is overwritten in place (each block owns its rows).
// ---------------------------------------------------------------------------
__global__ __launch_bounds__(256) void mlp_kernel(
    const float* __restrict__ x, float* __restrict__ x1out,
    const float* __restrict__ cbias, const float* __restrict__ gamma,
    const float* __restrict__ beta, const __bf16* __restrict__ W1b,
    const float* __restrict__ b1, const __bf16* __restrict__ W2b,
    const float* __restrict__ b2, const float* __restrict__ lsc, int N) {
  __shared__ __align__(16) __bf16 sW1[HH * CCH];   // 32KB: W1 bf16 [h][c]
  __shared__ __align__(16) __bf16 sT[8][16 * 64];  // 16KB: per-wave staging

  int lane = threadIdx.x & 31, wave = threadIdx.x >> 5;
  int lo = lane & 15, hi = lane >> 4;
  int rbase = blockIdx.x * 128 + wave * 16;

  // --- TDM: async DMA of W1b (256x64 bf16 tile) into LDS ---
  if (threadIdx.x == 0) {
    unsigned lds_addr = (unsigned)(size_t)(void*)&sW1[0];
    unsigned long long ga = (unsigned long long)(size_t)W1b;
    v4u g0 = { 1u,                                   // count=1
               lds_addr,                             // lds_addr
               (unsigned)(ga & 0xffffffffu),         // global_addr lo
               (unsigned)((ga >> 32) & 0x1ffffffu) | (2u << 30) };  // hi | type=2
    v8i g1 = { (int)(1u << 16),     // data_size=1 (2 bytes/elem)
               (int)(64u << 16),    // tensor_dim0 = 64 (bits 63:48)
               (int)(256u << 16),   // tensor_dim1 = 256 (bits 95:80)
               (int)(64u << 16),    // tile_dim0 = 64 (bits 127:112)
               (int)256,            // tile_dim1 = 256, tile_dim2 = 0
               (int)64,             // tensor_dim0_stride = 64
               0, 0 };
    v4i gz = { 0, 0, 0, 0 };
#if defined(__clang_major__) && (__clang_major__ >= 23)
    v8i gz8 = { 0, 0, 0, 0, 0, 0, 0, 0 };
    __builtin_amdgcn_tensor_load_to_lds(g0, g1, gz, gz, gz8, 0);
#else
    __builtin_amdgcn_tensor_load_to_lds(g0, g1, gz, gz, 0);
#endif
  }
  __builtin_amdgcn_s_wait_tensorcnt(0);
  __syncthreads();

  // --- bias + LayerNorm, producing A fragments directly in ISA layout ---
  int r = rbase + lo;
  int rc = (r < N) ? r : (N - 1);
  const float* row = x1out + (long long)rc * CCH;
  int k0 = hi * 8;
  float va[32];
#pragma unroll
  for (int g = 0; g < 4; g++) {
    v4f u0 = *(const v4f*)(row + k0 + g * 16);
    v4f u1 = *(const v4f*)(row + k0 + g * 16 + 4);
#pragma unroll
    for (int t = 0; t < 4; t++) {
      va[g * 8 + t]     = u0[t] + cbias[k0 + g * 16 + t];
      va[g * 8 + 4 + t] = u1[t] + cbias[k0 + g * 16 + 4 + t];
    }
  }
  float s = 0.f, q = 0.f;
#pragma unroll
  for (int i = 0; i < 32; i++) { s += va[i]; q += va[i] * va[i]; }
  s += __shfl_xor(s, 16);     // partner half-lane holds the other 32 channels
  q += __shfl_xor(q, 16);
  float mu = s * (1.f / 64.f);
  float var = q * (1.f / 64.f) - mu * mu;
  float rinv = rsqrtf(var + 1e-5f);

  v16bf A0, A1;
#pragma unroll
  for (int g = 0; g < 4; g++) {
#pragma unroll
    for (int t = 0; t < 8; t++) {
      int c = k0 + g * 16 + t;
      float nv = (va[g * 8 + t] - mu) * rinv * gamma[c] + beta[c];
      if (g < 2) A0[g * 8 + t] = (__bf16)nv;
      else       A1[(g - 2) * 8 + t] = (__bf16)nv;
    }
  }

  v8f acc2[4];
#pragma unroll
  for (int ct = 0; ct < 4; ct++) { v8f z = {}; acc2[ct] = z; }

  __bf16* st = &sT[wave][0];

  // --- 4 chunks of 64 hidden units: GEMM1 + GELU -> stage -> GEMM2 partial ---
  for (int chunk = 0; chunk < 4; chunk++) {
#pragma unroll
    for (int nt = 0; nt < 4; nt++) {
      int h = (chunk * 4 + nt) * 16 + lo;
      const __bf16* w1r = &sW1[h * CCH];
      float bb = b1[h];
      v8f c;
#pragma unroll
      for (int j = 0; j < 8; j++) c[j] = bb;
      v16bf B0 = frag_bf16(w1r + hi * 8,      w1r + hi * 8 + 16);
      v16bf B1 = frag_bf16(w1r + 32 + hi * 8, w1r + 32 + hi * 8 + 16);
      c = __builtin_amdgcn_wmma_f32_16x16x32_bf16(false, A0, false, B0,
                                                  (short)0, c, false, false);
      c = __builtin_amdgcn_wmma_f32_16x16x32_bf16(false, A1, false, B1,
                                                  (short)0, c, false, false);
#pragma unroll
      for (int j = 0; j < 8; j++) {
        float v = c[j];
        float gl = 0.5f * v * (1.0f + erff(v * 0.70710678f));  // exact GELU
        int m = j + 8 * hi;
        st[m * 64 + nt * 16 + lo] = (__bf16)gl;  // row-major [m][h_local]
      }
    }
    // consume this chunk's 64 K-values into the second GEMM
#pragma unroll
    for (int kf = 0; kf < 2; kf++) {
      int w = kf * 32;
      v16bf A2 = frag_bf16(st + lo * 64 + w + hi * 8,
                           st + lo * 64 + w + hi * 8 + 16);
      int kglob = chunk * 64 + w + hi * 8;
#pragma unroll
      for (int ct = 0; ct < 4; ct++) {
        const __bf16* w2r = W2b + (ct * 16 + lo) * HH + kglob;  // L2-resident
        v16bf B2 = frag_bf16(w2r, w2r + 16);
        acc2[ct] = __builtin_amdgcn_wmma_f32_16x16x32_bf16(
            false, A2, false, B2, (short)0, acc2[ct], false, false);
      }
    }
  }

  // --- bias2 + layer_scale + residual, write in place ---
#pragma unroll
  for (int ct = 0; ct < 4; ct++) {
    int c = ct * 16 + lo;
    float b2v = b2[c], lsv = lsc[c];
#pragma unroll
    for (int j = 0; j < 8; j++) {
      int rr = rbase + j + 8 * hi;
      if (rr < N) {
        float hv = acc2[ct][j] + b2v;
        x1out[(long long)rr * CCH + c] = lsv * hv + x[(long long)rr * CCH + c];
      }
    }
  }
}

extern "C" void kernel_launch(void* const* d_in, const int* in_sizes, int n_in,
                              void* d_out, int out_size, void* d_ws, size_t ws_size,
                              hipStream_t stream) {
  const float*     x     = (const float*)d_in[0];
  const float*     basis = (const float*)d_in[1];
  /* d_in[2] fiber_kernel_basis unused */
  const long long* ei    = (const long long*)d_in[3];
  const float*     Wk    = (const float*)d_in[4];
  const float*     cbias = (const float*)d_in[5];
  const float*     gamma = (const float*)d_in[6];
  const float*     beta  = (const float*)d_in[7];
  const float*     W1    = (const float*)d_in[8];
  const float*     b1    = (const float*)d_in[9];
  const float*     W2    = (const float*)d_in[10];
  const float*     b2    = (const float*)d_in[11];
  const float*     lsc   = (const float*)d_in[12];

  int N  = in_sizes[0] / CCH;
  int E  = in_sizes[3] / 2;
  int n1 = in_sizes[8];    // H*C
  int n2 = in_sizes[10];   // C*H

  __bf16* W1b = (__bf16*)d_ws;
  __bf16* W2b = W1b + n1;

  float* x1 = (float*)d_out;  // scatter accumulator lives in d_out
  hipMemsetAsync(d_out, 0, (size_t)N * CCH * sizeof(float), stream);

  int nmax = n1 > n2 ? n1 : n2;
  cvt_kernel<<<(nmax + 255) / 256, 256, 0, stream>>>(W1, W2, W1b, W2b, n1, n2);
  edge_kernel<<<(E + 127) / 128, 256, 0, stream>>>(x, basis, ei, Wk, x1, E);
  mlp_kernel<<<(N + 127) / 128, 256, 0, stream>>>(x, x1, cbias, gamma, beta,
                                                  W1b, b1, W2b, b2, lsc, N);
}
